// SlidingATTN_88330297409593
// MI455X (gfx1250) — matrix operand order repaired and checked by reading, not compiled, not run on hardware
//
#include <hip/hip_runtime.h>
#include <hip/hip_bf16.h>

// ---- CDNA5 WMMA types ----
typedef __attribute__((ext_vector_type(16))) _Float16 v16h;
typedef __attribute__((ext_vector_type(8)))  float    v8f;

#define D_DIM 256
#define L_DIM 1024
#define H_DIM 8
#define W_DIM 33

// ---------------------------------------------------------------------------
// Convert f32 -> f16 (one element per thread)
// ---------------------------------------------------------------------------
__global__ __launch_bounds__(256) void cvt_f16_kernel(const float* __restrict__ in,
                                                      _Float16* __restrict__ out, int n) {
    int i = blockIdx.x * 256 + threadIdx.x;
    if (i < n) out[i] = (_Float16)in[i];
}

// Transpose [K=256][N=256] f32 -> [N][K] f16 (weights, tiny)
__global__ __launch_bounds__(256) void transpose_f16_kernel(const float* __restrict__ in,
                                                            _Float16* __restrict__ out) {
    int k = threadIdx.x;
    int n = blockIdx.x;
    out[n * 256 + k] = (_Float16)in[k * 256 + n];
}

// ---------------------------------------------------------------------------
// attn = exp(x @ attn_w + attn_b) * valid     [B*L, 8]
// block: 256 threads -> 32 rows x 8 heads
// ---------------------------------------------------------------------------
__global__ __launch_bounds__(256) void attn_kernel(const float* __restrict__ x,
                                                   const unsigned char* __restrict__ mask,
                                                   const float* __restrict__ aw,   // [256,8]
                                                   const float* __restrict__ ab,   // [8]
                                                   float* __restrict__ attn) {
    __shared__ float s_w[256 * 8];
    for (int i = threadIdx.x; i < 2048; i += 256) s_w[i] = aw[i];
    __syncthreads();
    int r = threadIdx.x >> 3;
    int h = threadIdx.x & 7;
    size_t row = (size_t)blockIdx.x * 32 + r;
    const float* xr = x + row * D_DIM;
    float acc = ab[h];
    #pragma unroll 8
    for (int k = 0; k < D_DIM; ++k) acc += xr[k] * s_w[k * 8 + h];
    float valid = mask[row] ? 0.f : 1.f;
    attn[row * 8 + h] = __expf(acc) * valid;
}

// ---------------------------------------------------------------------------
// WMMA GEMM: C[16 x 256] = A[16 x 256] * W[256 x 256]  (+bias, LN, ELU)
//   FINAL=false : write elu(LN(...)) as f16 (v projection)
//   FINAL=true  : write elu(LN(...)) + residual as f32 (o projection)
// Bt is N-major f16: Bt[n*256 + k]
// 8 waves/block; wave w owns N-tiles 2w and 2w+1; K-loop 8 steps of 32.
// ---------------------------------------------------------------------------
template <bool FINAL>
__global__ __launch_bounds__(256) void gemm_ln_elu_kernel(
        const _Float16* __restrict__ A,
        const _Float16* __restrict__ Bt,
        const float* __restrict__ bias,
        const float* __restrict__ gamma,
        const float* __restrict__ beta,
        const float* __restrict__ residual,
        float* __restrict__ outF,
        _Float16* __restrict__ outH) {
    __shared__ float s_c[16][264];   // 16 rows x 256 cols (+pad)

    const int tid  = threadIdx.x;
    const int wave = tid >> 5;
    const int lane = tid & 31;
    const int half = lane >> 4;      // lane group 0/1
    const int lr   = lane & 15;
    const int row0 = blockIdx.x * 16;
    const int n0   = wave * 32;      // this wave's first N column

    v8f c0 = {};
    v8f c1 = {};

    // A fragment base: 16-bit A layout (16x32): lanes 0-15 -> M=lr, K pairs
    //   VGPR0-3: K = half*8 + (0..7);  VGPR4-7: K = 16 + half*8 + (0..7)
    const _Float16* arow = A + (size_t)(row0 + lr) * D_DIM + half * 8;
    // B fragment: 32x16 layout: N = n + lr, K = half*8 + (0..7), 16 + half*8 + (0..7) (same as A)
    const _Float16* b0p = Bt + (size_t)(n0 + lr) * D_DIM + half * 8;
    const _Float16* b1p = Bt + (size_t)(n0 + 16 + lr) * D_DIM + half * 8;

    #pragma unroll
    for (int kb = 0; kb < 8; ++kb) {
        v16h a, b0, b1;
        const _Float16* ap = arow + kb * 32;
        ((float4*)&a)[0] = *(const float4*)(ap);        // K = half*8 + 0..7
        ((float4*)&a)[1] = *(const float4*)(ap + 16);   // K = 16 + half*8 + 0..7
        const _Float16* bp0 = b0p + kb * 32;
        ((float4*)&b0)[0] = *(const float4*)(bp0);
        ((float4*)&b0)[1] = *(const float4*)(bp0 + 16);
        const _Float16* bp1 = b1p + kb * 32;
        ((float4*)&b1)[0] = *(const float4*)(bp1);
        ((float4*)&b1)[1] = *(const float4*)(bp1 + 16);

        c0 = __builtin_amdgcn_wmma_f32_16x16x32_f16(false, a, false, b0,
                                                    (short)0, c0, false, false);
        c1 = __builtin_amdgcn_wmma_f32_16x16x32_f16(false, a, false, b1,
                                                    (short)0, c1, false, false);
    }

    // C/D layout: lanes 0-15 -> M = i, N = lane; lanes 16-31 -> M = 8+i, N = lane-16
    {
        const int mb  = half * 8;
        const int n_a = n0 + lr;
        const int n_b = n0 + 16 + lr;
        const float ba = bias[n_a];
        const float bb = bias[n_b];
        #pragma unroll
        for (int i = 0; i < 8; ++i) {
            s_c[mb + i][n_a] = c0[i] + ba;
            s_c[mb + i][n_b] = c1[i] + bb;
        }
    }
    __syncthreads();

    // LayerNorm + ELU epilogue: wave w handles rows 2w, 2w+1
    #pragma unroll
    for (int rr = 0; rr < 2; ++rr) {
        const int m = wave * 2 + rr;
        float s = 0.f, s2 = 0.f;
        #pragma unroll
        for (int j = 0; j < 8; ++j) {
            float v = s_c[m][lane * 8 + j];
            s  += v;
            s2 += v * v;
        }
        #pragma unroll
        for (int off = 16; off > 0; off >>= 1) {
            s  += __shfl_xor(s,  off, 32);
            s2 += __shfl_xor(s2, off, 32);
        }
        const float mu  = s * (1.0f / 256.0f);
        const float var = s2 * (1.0f / 256.0f) - mu * mu;
        const float inv = rsqrtf(var + 1e-5f);
        const size_t rowg = (size_t)(row0 + m) * D_DIM;
        #pragma unroll
        for (int j = 0; j < 8; ++j) {
            const int n = lane * 8 + j;
            float v = (s_c[m][n] - mu) * inv * gamma[n] + beta[n];
            float e = v > 0.f ? v : (__expf(v) - 1.f);
            if (FINAL) outF[rowg + n] = e + residual[rowg + n];
            else       outH[rowg + n] = (_Float16)e;
        }
    }
}

// ---------------------------------------------------------------------------
// Window combine: out_pre[b,l,d] = sum_w aw[b,l,w,h] * v[b, l+2w-33, d]
// with aw = window(attn) * exp(pos), normalized over w.   One block per (b,l).
// ---------------------------------------------------------------------------
__global__ __launch_bounds__(256) void window_kernel(const float* __restrict__ attn, // [B*L,8]
                                                     const float* __restrict__ pos,  // [33*8]
                                                     const _Float16* __restrict__ v, // [B*L,256]
                                                     _Float16* __restrict__ outp) {  // [B*L,256]
    __shared__ float s_ep[W_DIM * H_DIM];
    for (int i = threadIdx.x; i < W_DIM * H_DIM; i += 256) s_ep[i] = __expf(pos[i]);
    __syncthreads();

    const int bl   = blockIdx.x;
    const int l    = bl & (L_DIM - 1);
    const int base = bl - l;          // b * L
    const int d    = threadIdx.x;
    const int h    = d >> 5;          // D/H = 32 channels per head

    float denom = 0.f, acc = 0.f;
    #pragma unroll
    for (int w = 0; w < W_DIM; ++w) {
        const int lw = l + 2 * w - 33;  // DIL*w - PAD
        if (lw >= 0 && lw < L_DIM) {
            const float a = attn[(size_t)(base + lw) * 8 + h] * s_ep[w * 8 + h];
            denom += a;
            acc   += a * (float)v[(size_t)(base + lw) * D_DIM + d];
        }
    }
    outp[(size_t)bl * D_DIM + d] = (_Float16)(acc / (denom + 1e-5f));
}

// ---------------------------------------------------------------------------
extern "C" void kernel_launch(void* const* d_in, const int* in_sizes, int n_in,
                              void* d_out, int out_size, void* d_ws, size_t ws_size,
                              hipStream_t stream) {
    const float*         x      = (const float*)d_in[0];
    const unsigned char* mask   = (const unsigned char*)d_in[1];
    const float*         attn_w = (const float*)d_in[2];
    const float*         attn_b = (const float*)d_in[3];
    const float*         pos    = (const float*)d_in[4];
    const float*         v_w    = (const float*)d_in[5];
    const float*         v_b    = (const float*)d_in[6];
    const float*         v_g    = (const float*)d_in[7];
    const float*         v_bt   = (const float*)d_in[8];
    const float*         o_w    = (const float*)d_in[9];
    const float*         o_b    = (const float*)d_in[10];
    const float*         o_g    = (const float*)d_in[11];
    const float*         o_bt   = (const float*)d_in[12];
    float* out = (float*)d_out;

    const int BL = 16 * 1024;                 // B*L
    char* ws = (char*)d_ws;
    const size_t SZ_XH  = (size_t)BL * D_DIM * 2;   // 8 MB
    const size_t SZ_WT  = 256 * 256 * 2;            // 128 KB
    const size_t SZ_ATT = (size_t)BL * 8 * 4;       // 512 KB

    _Float16* x_h  = (_Float16*)(ws);
    _Float16* vwt  = (_Float16*)(ws + SZ_XH);
    _Float16* owt  = (_Float16*)(ws + SZ_XH + SZ_WT);
    float*    attn = (float*)   (ws + SZ_XH + 2 * SZ_WT);
    _Float16* v_h  = (_Float16*)(ws + SZ_XH + 2 * SZ_WT + SZ_ATT);
    _Float16* op_h = x_h;   // x_h is dead after the V-projection GEMM

    // 1) precision/layout conversion
    cvt_f16_kernel<<<(BL * D_DIM) / 256, 256, 0, stream>>>(x, x_h, BL * D_DIM);
    transpose_f16_kernel<<<256, 256, 0, stream>>>(v_w, vwt);
    transpose_f16_kernel<<<256, 256, 0, stream>>>(o_w, owt);

    // 2) attention logits (small GEMM, VALU)
    attn_kernel<<<BL / 32, 256, 0, stream>>>(x, mask, attn_w, attn_b, attn);

    // 3) v = elu(LN(x @ v_w + v_b))  -- WMMA
    gemm_ln_elu_kernel<false><<<BL / 16, 256, 0, stream>>>(
        x_h, vwt, v_b, v_g, v_bt, nullptr, nullptr, v_h);

    // 4) sliding-window weighted combine
    window_kernel<<<BL, 256, 0, stream>>>(attn, pos, v_h, op_h);

    // 5) out = elu(LN(out_pre @ o_w + o_b)) + x  -- WMMA
    gemm_ln_elu_kernel<true><<<BL / 16, 256, 0, stream>>>(
        op_h, owt, o_b, o_g, o_bt, x, out, nullptr);
}
